// HGNN_lstm_15522011808194
// MI455X (gfx1250) — compile-verified, matrix operand order checked by reading, not run on hardware
//
#include <hip/hip_runtime.h>
#include <hip/hip_bf16.h>
#include <math.h>

// Problem constants (match reference)
#define B_   8
#define T_   32
#define N_   24
#define D_   128
#define C_   6
#define BT_  (B_ * T_)            // 256
#define BTN_ (B_ * T_ * N_)       // 6144
#define BTNN_ (B_ * T_ * N_ * N_) // 147456

typedef __bf16 bf16;
typedef __attribute__((ext_vector_type(16))) __bf16 v16bf;
typedef __attribute__((ext_vector_type(8)))  __bf16 v8bf;
typedef __attribute__((ext_vector_type(8)))  float  v8f;
typedef int v4i __attribute__((ext_vector_type(4)));
typedef v4i __attribute__((address_space(1)))* gv4i_p;   // global 16B vector ptr
typedef v4i __attribute__((address_space(3)))* lv4i_p;   // LDS 16B vector ptr

// ---------------------------------------------------------------------------
// WMMA helpers (wave32, 16x16 tiles, K stepped by 32, bf16 -> f32 accumulate)
// ---------------------------------------------------------------------------
__device__ __forceinline__ v8f wmma_bf16(v16bf a, v16bf b, v8f c) {
  return __builtin_amdgcn_wmma_f32_16x16x32_bf16(
      /*neg_a=*/false, a, /*neg_b=*/false, b,
      /*c_mod=*/(short)0, c, /*reuse_a=*/false, /*reuse_b=*/false);
}

__device__ __forceinline__ v16bf combine16(v8bf lo, v8bf hi) {
  return __builtin_shufflevector(lo, hi, 0, 1, 2, 3, 4, 5, 6, 7,
                                 8, 9, 10, 11, 12, 13, 14, 15);
}

// A tile 16x32 from row-major [*,128] source; two 16B vector loads per lane.
// lanes 0-15: M=lane, K-span base 0 ; lanes 16-31: M=lane-16, base +8
__device__ __forceinline__ v16bf load_frag_a(const bf16* __restrict__ A, int lane) {
  const int m = lane & 15, half = lane >> 4;
  const bf16* p = A + m * 128 + half * 8;
  return combine16(*(const v8bf*)p, *(const v8bf*)(p + 16));
}

// B fragment from pre-packed weights: [ntile][ktile][lane][16] contiguous.
__device__ __forceinline__ v16bf load_frag_bp(const bf16* __restrict__ Bp,
                                              int nt, int kt, int lane) {
  return *(const v16bf*)(Bp + (((nt * 4 + kt) * 32 + lane) << 4));
}

__device__ __forceinline__ float sigmoidf_(float x) { return 1.f / (1.f + expf(-x)); }

// ---------------------------------------------------------------------------
// Pack weight W[O=Nc, I=128] (f32 row-major) into WMMA B-fragment order (bf16).
// ---------------------------------------------------------------------------
__global__ void wpack_k(const float* __restrict__ W, bf16* __restrict__ Bp, int O) {
  const int idx = blockIdx.x * 256 + threadIdx.x;
  if (idx >= (O / 16) * 4 * 32) return;
  const int lane = idx & 31;
  const int kt = (idx >> 5) & 3;
  const int nt = idx >> 7;
  const int n = lane & 15, half = lane >> 4;
  const int col = nt * 16 + n;              // output-channel index o
  const int k0 = kt * 32 + half * 8;        // K base for this lane
  const float* src = W + col * 128;
  bf16* dst = Bp + ((size_t)idx << 4);
#pragma unroll
  for (int j = 0; j < 8; ++j)  dst[j] = (bf16)src[k0 + j];
#pragma unroll
  for (int j = 8; j < 16; ++j) dst[j] = (bf16)src[k0 + 8 + j];
}

// ---------------------------------------------------------------------------
// GEMM: C[M,NC] = A[M,128] x B[128,NC]; B pre-packed; NC compile-time so the
// epilogue is one base address + immediate-offset store clause.
// Each wave: 16 rows x (NT*16) cols, NT accumulators sharing one A fragment.
// ---------------------------------------------------------------------------
template <int NC, int NT>
__global__ void gemm_bf16_k(const bf16* __restrict__ A, const bf16* __restrict__ Bp,
                            float* __restrict__ C) {
  constexpr int NGRP = (NC / 16) / NT;
  const int lane = threadIdx.x & 31;
  const int wave = threadIdx.x >> 5;
  const int task = blockIdx.x * 8 + wave;
  const int mt = task / NGRP, ng = task % NGRP;
  const int nt0 = ng * NT;
  const bf16* Abase = A + mt * (16 * 128);
  __builtin_prefetch(Abase + 16 * 128, 0, 1);       // next M stripe (streaming)
  v8f acc[NT];
#pragma unroll
  for (int j = 0; j < NT; ++j) acc[j] = (v8f){};
#pragma unroll
  for (int kt = 0; kt < 4; ++kt) {
    const v16bf a = load_frag_a(Abase + kt * 32, lane);
#pragma unroll
    for (int j = 0; j < NT; ++j) {
      const v16bf b = load_frag_bp(Bp, nt0 + j, kt, lane);
      acc[j] = wmma_bf16(a, b, acc[j]);
    }
  }
  const int n = lane & 15, mhi = (lane >> 4) * 8;
  float* C0 = C + (mt * 16 + mhi) * NC + nt0 * 16 + n;
#pragma unroll
  for (int v = 0; v < 8; ++v)
#pragma unroll
    for (int j = 0; j < NT; ++j)
      C0[v * NC + j * 16] = acc[j][v];                // immediate offsets
}

// ---------------------------------------------------------------------------
// link_fun fused: hid = relu(W1 @ e + b1); adj = W2 . hid + b2; gate = sigm*mask
// One block (8 waves) per 16 edges; the 16x128 A tile is staged once in LDS
// via gfx1250 async global->LDS DMA (row pad 8 bf16 -> conflict-free
// ds_load_b128 fragment reads), 8x global-traffic reduction.
// ---------------------------------------------------------------------------
#define APITCH 136   // 128 + 8 bf16 pad (272B row -> rotates LDS banks)

__global__ void link_k(const bf16* __restrict__ hedge, const bf16* __restrict__ W1p,
                       const float* __restrict__ b1, const float* __restrict__ W2,
                       const float* __restrict__ b2, const int* __restrict__ nnum,
                       float* __restrict__ gate) {
  __shared__ bf16 atile[16 * APITCH];
  __shared__ float adj[16];
  const int tid = threadIdx.x;
  const int lane = tid & 31;
  const int wave = tid >> 5;
  if (tid < 16) adj[tid] = 0.f;

  const int e0 = blockIdx.x * 16;
  // cooperative stage: thread tid moves 8 bf16 (16B); row=tid>>4, chunk=tid&15
  {
    const int row = tid >> 4, chunk = tid & 15;
    bf16* dst = atile + row * APITCH + chunk * 8;
    const bf16* src = hedge + e0 * 128 + tid * 8;
#if __has_builtin(__builtin_amdgcn_global_load_async_to_lds_b128) && \
    __has_builtin(__builtin_amdgcn_s_wait_asynccnt)
    __builtin_amdgcn_global_load_async_to_lds_b128(
        (gv4i_p)(bf16*)src, (lv4i_p)dst, 0, 0);
    __builtin_amdgcn_s_wait_asynccnt(0);
#else
    *(v8bf*)dst = *(const v8bf*)src;
#endif
  }
  __syncthreads();

  v8f acc = {};
#pragma unroll
  for (int kt = 0; kt < 4; ++kt) {
    const bf16* pa = atile + (lane & 15) * APITCH + (lane >> 4) * 8 + kt * 32;
    const v16bf a = combine16(*(const v8bf*)pa, *(const v8bf*)(pa + 16)); // ds_load_b128 x2
    const v16bf b = load_frag_bp(W1p, wave, kt, lane);
    acc = wmma_bf16(a, b, acc);
  }
  const int col = wave * 16 + (lane & 15);
  const int mhi = (lane >> 4) * 8;
  const float w2 = W2[col], bb = b1[col];
#pragma unroll
  for (int v = 0; v < 8; ++v) {
    const float x = fmaxf(acc[v] + bb, 0.f) * w2;
    atomicAdd(&adj[mhi + v], x);
  }
  __syncthreads();
  if (tid < 16) {
    const int e = e0 + tid;
    const int w = e % N_;
    const int v = (e / N_) % N_;
    const int bt = e / (N_ * N_);
    const int nn = nnum[bt];
    float g = 0.f;
    if (v < nn && w < nn) g = sigmoidf_(adj[tid] + b2[0]);
    gate[e] = g;
  }
}

// ---------------------------------------------------------------------------
// Message + gate + h_edge update + sender-sum (m_v). One block per (b,t,v).
// ---------------------------------------------------------------------------
__global__ void msg_k(const float* __restrict__ WhH, const float* __restrict__ WeE,
                      const float* __restrict__ gate, const float* __restrict__ msg_b,
                      const bf16* __restrict__ edgeT, const int* __restrict__ nnum,
                      bf16* __restrict__ hedge, bf16* __restrict__ mv_h) {
  const int r = blockIdx.x;            // (b*T+t)*N + v
  const int d = threadIdx.x;           // 0..127
  const int v = r % N_;
  const int bt = r / N_;
  const int nn = nnum[bt];
  const bool vin = v < nn;
  const float mb = msg_b[d];
  float acc = 0.f;
#pragma unroll 4
  for (int w = 0; w < N_; ++w) {
    const int e = r * N_ + w;
    const float g = gate[e];
    float x = WhH[(bt * N_ + w) * 128 + d] + WeE[(size_t)e * 128 + d] + mb;
    x = fmaxf(x, 0.f);
    const float gated = g * x;
    const bool pin = vin && (w < nn);
    hedge[(size_t)e * 128 + d] = pin ? (bf16)gated : edgeT[(size_t)e * 128 + d];
    acc += gated;
  }
  mv_h[r * 128 + d] = (bf16)acc;
}

// ---------------------------------------------------------------------------
// GRU elementwise update (gi/gh from WMMA GEMMs). One block per node row.
// ---------------------------------------------------------------------------
__global__ void gru_k(const float* __restrict__ gi, const float* __restrict__ gh,
                      const float* __restrict__ bih, const float* __restrict__ bhh,
                      const float* __restrict__ noderesT, const int* __restrict__ nnum,
                      float* __restrict__ hnode, bf16* __restrict__ hnode_h) {
  const int r = blockIdx.x, d = threadIdx.x;
  const int gb = r * 384;
  const float ir  = gi[gb + d]       + bih[d],       hr = gh[gb + d]       + bhh[d];
  const float iz  = gi[gb + 128 + d] + bih[128 + d], hz = gh[gb + 128 + d] + bhh[128 + d];
  const float in_ = gi[gb + 256 + d] + bih[256 + d], hn = gh[gb + 256 + d] + bhh[256 + d];
  const float rg = sigmoidf_(ir + hr);
  const float z  = sigmoidf_(iz + hz);
  const float n  = tanhf(in_ + rg * hn);
  const int hi = r * 128 + d;
  const float hprev = hnode[hi];
  const float hnew = (1.f - z) * n + z * hprev;
  const bool m = (r % N_) < nnum[r / N_];
  const float outv = m ? hnew : noderesT[hi];
  hnode[hi] = outv;
  hnode_h[hi] = (bf16)outv;
}

// ---------------------------------------------------------------------------
// LSTM cell for one time step (hg from WMMA GEMM on previous h).
// ---------------------------------------------------------------------------
__global__ void lstm_cell_k(const float* __restrict__ xg, const float* __restrict__ hg,
                            const float* __restrict__ bih, const float* __restrict__ bhh,
                            float* __restrict__ cstate, float* __restrict__ hstate,
                            bf16* __restrict__ hstate_h, float* __restrict__ lstm_out,
                            int t) {
  const int rr = blockIdx.x;          // b*N + n   (LSTM batch row)
  const int d = threadIdx.x;
  const int b = rr / N_, n = rr % N_;
  const int xrow = (b * T_ + t) * N_ + n;
  const int xb = xrow * 512, hb = rr * 512;
  const float gi = xg[xb + d]       + hg[hb + d]       + bih[d]       + bhh[d];
  const float gf = xg[xb + 128 + d] + hg[hb + 128 + d] + bih[128 + d] + bhh[128 + d];
  const float gg = xg[xb + 256 + d] + hg[hb + 256 + d] + bih[256 + d] + bhh[256 + d];
  const float go = xg[xb + 384 + d] + hg[hb + 384 + d] + bih[384 + d] + bhh[384 + d];
  const int si = rr * 128 + d;
  const float c = sigmoidf_(gf) * cstate[si] + sigmoidf_(gi) * tanhf(gg);
  const float h = sigmoidf_(go) * tanhf(c);
  cstate[si] = c;
  hstate[si] = h;
  hstate_h[si] = (bf16)h;
  lstm_out[xrow * 128 + d] = h;
}

// ---------------------------------------------------------------------------
// Data prep / utility kernels
// ---------------------------------------------------------------------------
__global__ void edge_prep_k(const float* __restrict__ src, bf16* __restrict__ edgeT,
                            bf16* __restrict__ hedge) {
  const long i = (long)blockIdx.x * 256 + threadIdx.x;   // over BTNN*128
  if (i >= (long)BTNN_ * 128) return;
  const int d = (int)(i & 127);
  long r = i >> 7;
  const int w = (int)(r % N_); r /= N_;
  const int v = (int)(r % N_);
  const long bt = r / N_;
  const bf16 h = (bf16)src[(bt * 128 + d) * (N_ * N_) + v * N_ + w];
  edgeT[i] = h;
  hedge[i] = h;
}

__global__ void node_prep_k(const float* __restrict__ src, float* __restrict__ noderesT,
                            float* __restrict__ hnode, bf16* __restrict__ hnode_h) {
  const int i = blockIdx.x * 256 + threadIdx.x;          // over BTN*128
  if (i >= BTN_ * 128) return;
  const int d = i & 127;
  const int r = i >> 7;
  const int n = r % N_;
  const int bt = r / N_;
  const float val = src[(bt * 128 + d) * N_ + n];
  noderesT[i] = val;
  hnode[i] = val;
  hnode_h[i] = (bf16)val;
}

__global__ void lstm_init_k(float* __restrict__ h, bf16* __restrict__ hh,
                            float* __restrict__ c) {
  const int i = blockIdx.x * 256 + threadIdx.x;
  if (i >= (B_ * N_) * 128) return;
  h[i] = 0.f; hh[i] = (bf16)0.f; c[i] = 0.f;
}

__global__ void readout_k(const float* __restrict__ lstm_out, const float* __restrict__ roW,
                          const float* __restrict__ rob, const int* __restrict__ nnum,
                          float* __restrict__ out) {
  const int i = blockIdx.x * 256 + threadIdx.x;          // over BTN*C
  if (i >= BTN_ * C_) return;
  const int r = i / C_, c = i % C_;
  const int n = r % N_, bt = r / N_;
  float val = rob[c];
  const float* hv = lstm_out + r * 128;
  const float* wv = roW + c * 128;
#pragma unroll 8
  for (int d = 0; d < 128; ++d) val += hv[d] * wv[d];
  out[i] = (n < nnum[bt]) ? val : 0.f;
}

// ---------------------------------------------------------------------------
// Host launcher
// ---------------------------------------------------------------------------
extern "C" void kernel_launch(void* const* d_in, const int* in_sizes, int n_in,
                              void* d_out, int out_size, void* d_ws, size_t ws_size,
                              hipStream_t stream) {
  const float* node_resnet = (const float*)d_in[0];
  const float* edge_resnet = (const float*)d_in[1];
  const float* link_W1 = (const float*)d_in[2];
  const float* link_b1 = (const float*)d_in[3];
  const float* link_W2 = (const float*)d_in[4];
  const float* link_b2 = (const float*)d_in[5];
  const float* msg_Wh  = (const float*)d_in[6];
  const float* msg_We  = (const float*)d_in[7];
  const float* msg_b   = (const float*)d_in[8];
  const float* gru_Wih = (const float*)d_in[9];
  const float* gru_Whh = (const float*)d_in[10];
  const float* gru_bih = (const float*)d_in[11];
  const float* gru_bhh = (const float*)d_in[12];
  const float* lstm_Wih = (const float*)d_in[13];
  const float* lstm_Whh = (const float*)d_in[14];
  const float* lstm_bih = (const float*)d_in[15];
  const float* lstm_bhh = (const float*)d_in[16];
  const float* ro_W = (const float*)d_in[17];
  const float* ro_b = (const float*)d_in[18];
  const int*   nnum = (const int*)d_in[19];
  float* out = (float*)d_out;

  // Workspace carving (256B aligned)
  char* p = (char*)d_ws;
  auto alloc = [&](size_t bytes) -> void* {
    void* r = (void*)p;
    p += (bytes + 255) & ~(size_t)255;
    return r;
  };
  bf16*  edgeT   = (bf16*)alloc((size_t)BTNN_ * 128 * 2);
  bf16*  hedge   = (bf16*)alloc((size_t)BTNN_ * 128 * 2);
  float* WeE     = (float*)alloc((size_t)BTNN_ * 128 * 4);
  float* gate    = (float*)alloc((size_t)BTNN_ * 4);
  float* noderesT= (float*)alloc((size_t)BTN_ * 128 * 4);
  float* hnode   = (float*)alloc((size_t)BTN_ * 128 * 4);
  bf16*  hnode_h = (bf16*)alloc((size_t)BTN_ * 128 * 2);
  float* WhH     = (float*)alloc((size_t)BTN_ * 128 * 4);
  bf16*  mv_h    = (bf16*)alloc((size_t)BTN_ * 128 * 2);
  float* gi      = (float*)alloc((size_t)BTN_ * 384 * 4);
  float* gh      = (float*)alloc((size_t)BTN_ * 384 * 4);
  float* xg      = (float*)alloc((size_t)BTN_ * 512 * 4);
  float* hg      = (float*)alloc((size_t)(B_ * N_) * 512 * 4);
  float* hstate  = (float*)alloc((size_t)(B_ * N_) * 128 * 4);
  bf16*  hstate_h= (bf16*)alloc((size_t)(B_ * N_) * 128 * 2);
  float* cstate  = (float*)alloc((size_t)(B_ * N_) * 128 * 4);
  float* lstm_out= (float*)alloc((size_t)BTN_ * 128 * 4);
  bf16*  W1p     = (bf16*)alloc(128 * 128 * 2);   // packed fragment layouts
  bf16*  Whp     = (bf16*)alloc(128 * 128 * 2);
  bf16*  Wep     = (bf16*)alloc(128 * 128 * 2);
  bf16*  gWihp   = (bf16*)alloc(384 * 128 * 2);
  bf16*  gWhhp   = (bf16*)alloc(384 * 128 * 2);
  bf16*  lWihp   = (bf16*)alloc(512 * 128 * 2);
  bf16*  lWhhp   = (bf16*)alloc(512 * 128 * 2);

  auto gemm = [&](const bf16* A, const bf16* Bp, float* Cc, int M, int Nc) {
    if (Nc == 128)      gemm_bf16_k<128, 4><<<(M / 16) * 2 / 8, 256, 0, stream>>>(A, Bp, Cc);
    else if (Nc == 384) gemm_bf16_k<384, 4><<<(M / 16) * 6 / 8, 256, 0, stream>>>(A, Bp, Cc);
    else                gemm_bf16_k<512, 4><<<(M / 16) * 8 / 8, 256, 0, stream>>>(A, Bp, Cc);
  };
  auto pack = [&](const float* W, bf16* Bp, int O) {
    wpack_k<<<((O / 16) * 128 + 255) / 256, 256, 0, stream>>>(W, Bp, O);
  };

  // ---- prep: transposes + bf16 casts + weight fragment packing ------------
  edge_prep_k<<<(int)(((long)BTNN_ * 128 + 255) / 256), 256, 0, stream>>>(edge_resnet, edgeT, hedge);
  node_prep_k<<<(BTN_ * 128 + 255) / 256, 256, 0, stream>>>(node_resnet, noderesT, hnode, hnode_h);
  pack(link_W1, W1p, 128);
  pack(msg_Wh, Whp, 128);
  pack(msg_We, Wep, 128);
  pack(gru_Wih, gWihp, 384);
  pack(gru_Whh, gWhhp, 384);
  pack(lstm_Wih, lWihp, 512);
  pack(lstm_Whh, lWhhp, 512);

  // msg_We @ edge_resnet is loop-invariant: compute once (big WMMA GEMM)
  gemm(edgeT, Wep, WeE, BTNN_, 128);

  // ---- propagate layers ---------------------------------------------------
  for (int layer = 0; layer < 3; ++layer) {
    link_k<<<BTNN_ / 16, 256, 0, stream>>>(hedge, W1p, link_b1, link_W2, link_b2, nnum, gate);
    gemm(hnode_h, Whp, WhH, BTN_, 128);
    msg_k<<<BTN_, 128, 0, stream>>>(WhH, WeE, gate, msg_b, edgeT, nnum, hedge, mv_h);
    gemm(mv_h, gWihp, gi, BTN_, 384);
    gemm(hnode_h, gWhhp, gh, BTN_, 384);
    gru_k<<<BTN_, 128, 0, stream>>>(gi, gh, gru_bih, gru_bhh, noderesT, nnum, hnode, hnode_h);
  }

  // ---- temporal LSTM ------------------------------------------------------
  gemm(hnode_h, lWihp, xg, BTN_, 512);              // all x-gates up front
  lstm_init_k<<<((B_ * N_) * 128 + 255) / 256, 256, 0, stream>>>(hstate, hstate_h, cstate);
  for (int t = 0; t < T_; ++t) {
    gemm(hstate_h, lWhhp, hg, B_ * N_, 512);
    lstm_cell_k<<<B_ * N_, 128, 0, stream>>>(xg, hg, lstm_bih, lstm_bhh,
                                             cstate, hstate, hstate_h, lstm_out, t);
  }

  // ---- readout ------------------------------------------------------------
  readout_k<<<(BTN_ * C_ + 255) / 256, 256, 0, stream>>>(lstm_out, ro_W, ro_b, nnum, out);
}
